// self_attention_module_32298154066789
// MI455X (gfx1250) — matrix operand order repaired and checked by reading, not compiled, not run on hardware
//
#include <hip/hip_runtime.h>

typedef __bf16 bf16;
typedef __attribute__((ext_vector_type(16))) __bf16 bf16x16;
typedef __attribute__((ext_vector_type(8)))  float  f32x8;
typedef __attribute__((ext_vector_type(4)))  unsigned int u32x4;

constexpr int kB = 4;
constexpr int kC = 256;    // C_IN == C_TMP == C_OUT
constexpr int kN = 4096;

union FragU { bf16x16 v; u32x4 q[2]; };

__device__ __forceinline__ bf16 f2bf(float f) {
  union { float f; unsigned u; } c; c.f = f;
  unsigned u = c.u;
  unsigned r = (u + 0x7fffu + ((u >> 16) & 1u)) >> 16;   // round-to-nearest-even
  union { unsigned short s; bf16 b; } o; o.s = (unsigned short)r;
  return o.b;
}

// A-fragment (16x32, M x K): lane (r = lane&15, h = lane>>4) holds row M=r,
// K = h*16 .. h*16+15 contiguously (matches B-fragment order).
__device__ __forceinline__ bf16x16 load_a_frag(const bf16* rowk0, int h) {
  FragU u;
  u.q[0] = *(const u32x4*)(rowk0 + h * 16);
  u.q[1] = *(const u32x4*)(rowk0 + h * 16 + 8);
  return u.v;
}

// B-fragment (32x16, K x N): lane (j = lane&15, h) holds column j,
// K = h*16 .. h*16+15 contiguously (this code's order, same as load_a_frag).
__device__ __forceinline__ bf16x16 load_b_frag(const bf16* rowk0, int h) {
  FragU u;
  const u32x4* p = (const u32x4*)(rowk0 + h * 16);
  u.q[0] = p[0];
  u.q[1] = p[1];
  return u.v;
}

__device__ __forceinline__ f32x8 wmma_bf16(bf16x16 a, bf16x16 b, f32x8 c) {
  return __builtin_amdgcn_wmma_f32_16x16x32_bf16(false, a, false, b, (short)0, c,
                                                 false, false);
}

// CDNA5 async global->LDS copy (16B per lane), tracked by ASYNCcnt.
__device__ __forceinline__ void async_copy_b128(void* lds_dst, const void* gaddr) {
  unsigned lds = (unsigned)(unsigned long long)lds_dst;   // low 32b = LDS offset
  asm volatile("global_load_async_to_lds_b128 %0, %1, off"
               :: "v"(lds), "v"(gaddr) : "memory");
}
__device__ __forceinline__ void wait_async0() {
  asm volatile("s_wait_asynccnt 0x0" ::: "memory");
}

// ---------------- conversion kernels ----------------

__global__ void k_cvt_f32_bf16(const float* __restrict__ in, bf16* __restrict__ out, int n) {
  int i = blockIdx.x * blockDim.x + threadIdx.x;
  if (i < n) out[i] = f2bf(in[i]);
}

// feaT[b][n][c] = (bf16) fea[b][c][n]    (LDS 32x32 tile transpose)
__global__ void k_fea_transpose(const float* __restrict__ fea, bf16* __restrict__ feaT) {
  __shared__ float tile[32][33];
  int b  = blockIdx.z;
  int n0 = blockIdx.x * 32;
  int c0 = blockIdx.y * 32;
  int tx = threadIdx.x, ty = threadIdx.y;           // (32, 8)
  const float* f = fea + ((long long)b * kC + c0) * kN + n0;
#pragma unroll
  for (int i = 0; i < 4; i++)
    tile[ty + i * 8][tx] = f[(long long)(ty + i * 8) * kN + tx];
  __syncthreads();
  bf16* o = feaT + ((long long)b * kN + n0) * kC + c0;
#pragma unroll
  for (int i = 0; i < 4; i++)
    o[(long long)(ty + i * 8) * kC + tx] = f2bf(tile[tx][ty + i * 8]);
}

// ---------------- WMMA GEMM: C[m,n] = sum_k A[m,k] * Bt[n,k], bf16 in/out ----------------
// block = 256 threads (8 waves); each wave computes one 16x16 tile.
__global__ __launch_bounds__(256) void k_gemm_bf16(
    const bf16* __restrict__ A, const bf16* __restrict__ Bt, bf16* __restrict__ C,
    int M, int Nn, int K,
    long long aStride, long long bStride, long long cStride) {
  int b    = blockIdx.z;
  const bf16* Ab  = A  + (long long)b * aStride;
  const bf16* Btb = Bt + (long long)b * bStride;
  bf16*       Cb  = C  + (long long)b * cStride;

  int wave = threadIdx.x >> 5;
  int lane = threadIdx.x & 31;
  int r = lane & 15, h = lane >> 4;
  int m0 = blockIdx.x * 16;
  int n0 = blockIdx.y * 128 + wave * 16;

  f32x8 acc = {};
#pragma unroll
  for (int k0 = 0; k0 < kC; k0 += 32) {
    bf16x16 af = load_a_frag(Ab + (long long)(m0 + r) * K + k0, h);
    bf16x16 bf = load_b_frag(Btb + (long long)(n0 + r) * K + k0, h);
    acc = wmma_bf16(af, bf, acc);
  }
  // D layout: vgpr rr -> row m0+rr+8h, column n0 + (lane&15)
#pragma unroll
  for (int rr = 0; rr < 8; rr++)
    Cb[(long long)(m0 + rr + 8 * h) * Nn + n0 + r] = f2bf(acc[rr]);
}

// ---------------- fused attention (flash style, softmax over key axis) ----------------
// QT, KT: (B, N, C) bf16 ; V: (B, C, N) bf16 ; out: (B, 512, N) f32, channels [0,256)
// block = 128 threads (4 waves); each wave owns a 16-wide j tile, full o=256 range.
// Double-buffered LDS, filled by async global->LDS DMA overlapped with WMMA.
__global__ __launch_bounds__(128) void k_attn(
    const bf16* __restrict__ QT, const bf16* __restrict__ KT,
    const bf16* __restrict__ V, float* __restrict__ out) {
  __shared__ __align__(16) bf16 sK[2][32][kC];   // [buf][k_local][t]   2 x 16KB
  __shared__ __align__(16) bf16 sV[2][kC][32];   // [buf][o][k_local]   2 x 16KB

  int b     = blockIdx.x / (kN / 64);
  int j0blk = (blockIdx.x % (kN / 64)) * 64;
  int wave  = threadIdx.x >> 5;
  int lane  = threadIdx.x & 31;
  int r = lane & 15, h = lane >> 4;
  int jcol = j0blk + wave * 16 + r;           // this lane's output column

  const bf16* QTb = QT + (long long)b * kN * kC;
  const bf16* KTb = KT + (long long)b * kN * kC;
  const bf16* Vb  = V  + (long long)b * kC * kN;

  // issue async fill of K^T (32 x 256) and V (256 x 32) tiles for key block kb
  auto issue = [&](int kb, int buf) {
    for (int c = threadIdx.x; c < 1024; c += 128) {       // 8 x b128 per thread
      int row = c >> 5, col = (c & 31) << 3;
      async_copy_b128(&sK[buf][row][col],
                      KTb + (long long)(kb + row) * kC + col);
    }
    for (int c = threadIdx.x; c < 1024; c += 128) {       // 8 x b128 per thread
      int row = c >> 2, col = (c & 3) << 3;
      async_copy_b128(&sV[buf][row][col],
                      Vb + (long long)row * kN + kb + col);
    }
  };

  // cache Q B-fragments for all 8 t-steps (t contraction dim = 256)
  bf16x16 qf[8];
#pragma unroll
  for (int s = 0; s < 8; s++)
    qf[s] = load_b_frag(QTb + (long long)jcol * kC + s * 32, h);

  f32x8 acc[16] = {};                          // o = 0..255 in 16 tiles
  float m = -3.0e38f, l = 0.0f;

  constexpr int ITERS = kN / 32;               // 128 key blocks
  issue(0, 0);
  for (int it = 0; it < ITERS; it++) {
    int buf = it & 1;
    wait_async0();                             // my DMA for this buffer done
    __syncthreads();                           // everyone's DMA done; prev readers done
    if (it + 1 < ITERS) issue((it + 1) * 32, buf ^ 1);   // overlap next fill

    // S tiles: rows k (two 16-row tiles), cols j ; contraction over t=256
    f32x8 s0 = {}, s1 = {};
#pragma unroll
    for (int s = 0; s < 8; s++) {
      bf16x16 a0 = load_a_frag(&sK[buf][r][s * 32], h);
      bf16x16 a1 = load_a_frag(&sK[buf][16 + r][s * 32], h);
      s0 = wmma_bf16(a0, qf[s], s0);
      s1 = wmma_bf16(a1, qf[s], s1);
    }

    // online softmax over k (per column j; a lane holds only its own column)
    float tm = s0[0];
#pragma unroll
    for (int i = 0; i < 8; i++) { tm = fmaxf(tm, s0[i]); tm = fmaxf(tm, s1[i]); }
    tm = fmaxf(tm, __shfl_xor(tm, 16, 32));
    float mnew  = fmaxf(m, tm);
    float scale = __expf(m - mnew);
    m = mnew;
    l *= scale;
#pragma unroll
    for (int ot = 0; ot < 16; ot++) acc[ot] = acc[ot] * scale;

    float p0[8], p1[8], lsum = 0.0f;
#pragma unroll
    for (int i = 0; i < 8; i++) {
      p0[i] = __expf(s0[i] - mnew);
      p1[i] = __expf(s1[i] - mnew);
      lsum += p0[i] + p1[i];
    }
    l += lsum + __shfl_xor(lsum, 16, 32);

    // Re-pack exp(S) into a B-fragment (K = h*16 + pos). Lane(j,h=0) needs
    // k=0..15 (own tile0 rows 0..7 + partner's rows 8..15); lane(j,h=1) needs
    // k=16..31 (partner's tile1 rows 0..7 + own rows 8..15). One half-swap.
    FragU pf;
#pragma unroll
    for (int i = 0; i < 8; i++) {
      float o0 = __shfl_xor(p0[i], 16, 32);
      float o1 = __shfl_xor(p1[i], 16, 32);
      pf.v[i]     = f2bf(h == 0 ? p0[i] : o1);
      pf.v[i + 8] = f2bf(h == 0 ? o0    : p1[i]);
    }

    // acc += V_tile (16o x 32k) @ P (32k x 16j)
#pragma unroll
    for (int ot = 0; ot < 16; ot++) {
      bf16x16 vf = load_a_frag(&sV[buf][ot * 16 + r][0], h);
      acc[ot] = wmma_bf16(vf, pf.v, acc[ot]);
    }
  }

  // epilogue: divide by softmax denom, relu, write channels [0,256)
  float inv = 1.0f / l;
  float* outb = out + (long long)b * (2 * kC) * kN;
#pragma unroll
  for (int ot = 0; ot < 16; ot++)
#pragma unroll
    for (int rr = 0; rr < 8; rr++) {
      int o = ot * 16 + rr + 8 * h;
      float v = acc[ot][rr] * inv;
      outb[(long long)o * kN + jcol] = fmaxf(v, 0.0f);
    }
}

// channels [256,512) of out = relu(fea)
__global__ void k_relu_concat(const float* __restrict__ fea, float* __restrict__ out) {
  long long idx = (long long)blockIdx.x * blockDim.x + threadIdx.x;
  long long total = (long long)kB * kC * kN;
  if (idx >= total) return;
  long long b = idx / ((long long)kC * kN);
  out[idx + (b + 1) * (long long)kC * kN] = fmaxf(fea[idx], 0.0f);
}

// ---------------- launch ----------------

extern "C" void kernel_launch(void* const* d_in, const int* in_sizes, int n_in,
                              void* d_out, int out_size, void* d_ws, size_t ws_size,
                              hipStream_t stream) {
  const float* fea = (const float*)d_in[0];
  const float* Wq  = (const float*)d_in[1];
  const float* Wk  = (const float*)d_in[2];
  const float* Wv  = (const float*)d_in[3];
  float* out = (float*)d_out;

  char* ws = (char*)d_ws;
  const long long SZ = (long long)kB * kN * kC * 2;   // 8 MiB each
  bf16* feaT = (bf16*)(ws);
  bf16* QT   = (bf16*)(ws + SZ);
  bf16* KT   = (bf16*)(ws + 2 * SZ);
  bf16* Vb   = (bf16*)(ws + 3 * SZ);
  bf16* WqB  = (bf16*)(ws + 4 * SZ);
  bf16* WkB  = (bf16*)(ws + 4 * SZ + kC * kC * 2);
  bf16* WvB  = (bf16*)(ws + 4 * SZ + 2 * kC * kC * 2);

  // bf16 conversions
  k_cvt_f32_bf16<<<dim3((kC * kC + 255) / 256), 256, 0, stream>>>(Wq, WqB, kC * kC);
  k_cvt_f32_bf16<<<dim3((kC * kC + 255) / 256), 256, 0, stream>>>(Wk, WkB, kC * kC);
  k_cvt_f32_bf16<<<dim3((kC * kC + 255) / 256), 256, 0, stream>>>(Wv, WvB, kC * kC);
  k_fea_transpose<<<dim3(kN / 32, kC / 32, kB), dim3(32, 8), 0, stream>>>(fea, feaT);

  // QT[n,t] = sum_i feaT[n,i] * Wq[t,i]   (and same for KT)
  long long bnc = (long long)kN * kC;
  k_gemm_bf16<<<dim3(kN / 16, kC / 128, kB), 256, 0, stream>>>(
      feaT, WqB, QT, kN, kC, kC, bnc, 0, bnc);
  k_gemm_bf16<<<dim3(kN / 16, kC / 128, kB), 256, 0, stream>>>(
      feaT, WkB, KT, kN, kC, kC, bnc, 0, bnc);
  // V[o,n] = sum_i Wv[o,i] * feaT[n,i]
  k_gemm_bf16<<<dim3(kC / 16, kN / 128, kB), 256, 0, stream>>>(
      WvB, feaT, Vb, kC, kN, kC, 0, bnc, bnc);

  // fused K^T Q -> softmax(key axis) -> V @ P, writes out channels [0,256)
  k_attn<<<dim3(kB * (kN / 64)), 128, 0, stream>>>(QT, KT, Vb, out);

  // concat half: out channels [256,512) = relu(fea)
  long long total = (long long)kB * kC * kN;
  k_relu_concat<<<dim3((unsigned)((total + 255) / 256)), 256, 0, stream>>>(fea, out);
}
